// RecurrentMLPReservoir_42245298324087
// MI455X (gfx1250) — compile-verified, matrix operand order checked by reading, not run on hardware
//
#include <hip/hip_runtime.h>

#define INPUT_N 6
#define HIDDEN_N 50
#define OUTPUT_N 2
#define BB 4096
#define TT 2048

typedef __bf16 bf16_t;
typedef bf16_t v8bf  __attribute__((ext_vector_type(8)));
typedef bf16_t v16bf __attribute__((ext_vector_type(16)));
typedef float  v8f   __attribute__((ext_vector_type(8)));
typedef float  v4f   __attribute__((ext_vector_type(4)));

// f32 -> bf16 using the native converter (RTNE on gfx1250).
static __device__ __forceinline__ unsigned short f2bf(float f) {
  return __builtin_bit_cast(unsigned short, static_cast<bf16_t>(f));
}
static __device__ __forceinline__ float bf2f(unsigned short h) {
  unsigned u = ((unsigned)h) << 16;
  return __builtin_bit_cast(float, u);
}
static __device__ __forceinline__ float fast_tanh(float x) {
#if __has_builtin(__builtin_amdgcn_tanhf)
  return __builtin_amdgcn_tanhf(x);          // gfx1250 v_tanh_f32
#else
  float r;
  asm volatile("v_tanh_f32 %0, %1\n\tv_nop" : "=v"(r) : "v"(x));
  return r;
#endif
}

// Load one 16x32 bf16 WMMA fragment from a row-major [*,64] bf16 LDS tile.
// Lane l<16 holds row `row` K-chunks {0-7,16-23}; lane>=16 holds {8-15,24-31}.
static __device__ __forceinline__ v16bf load_frag(const unsigned short* m, int row,
                                                  int kb, int laneHi) {
  const char* p = (const char*)(m + row * 64 + kb * 32) + laneHi * 16;
  union { v16bf v; v8bf h[2]; } u;
  u.h[0] = *(const v8bf*)(p);        // ds_load_b128
  u.h[1] = *(const v8bf*)(p + 32);   // ds_load_b128
  return u.v;
}

__global__ __launch_bounds__(32)
void elman_wmma_kernel(const float* __restrict__ x, const float* __restrict__ h0,
                       const float* __restrict__ Wih_w, const float* __restrict__ Wih_b,
                       const float* __restrict__ Whh, const float* __restrict__ Who_w,
                       const float* __restrict__ Who_b, float* __restrict__ out) {
  // Wcat^T: [n=0..63][k=0..63] bf16, row-major. n<50: h-outputs; n=50,51: logits.
  __shared__ __align__(16) unsigned short sW[64 * 64];      // 8 KB
  // Augmented A rows: [row=0..15][k=0..63] bf16 = [h | x | 1 | 0]
  __shared__ __align__(16) unsigned short sA[16 * 64];      // 2 KB
  // Logit staging ring: [row][slot=0..63][o=0..1] f32
  __shared__ __align__(16) float sY[16 * 64 * 2];           // 8 KB

  const int lane = threadIdx.x;
  const int lm = lane & 15;
  const int lh = lane >> 4;
  const int row0 = blockIdx.x * 16;
  float* logits = out;
  float* hfin = out + (size_t)BB * TT * OUTPUT_N;

  // ---- Build Wcat^T in LDS (once) ----
  for (int idx = lane; idx < 64 * 64; idx += 32) {
    int n = idx >> 6, k = idx & 63;
    float v = 0.f;
    if (n < HIDDEN_N) {
      if (k < HIDDEN_N)                 v = Whh[k * HIDDEN_N + n];        // h @ W_hh
      else if (k < HIDDEN_N + INPUT_N)  v = Wih_w[n * INPUT_N + (k - HIDDEN_N)];
      else if (k == 56)                 v = Wih_b[n];                     // bias row
    } else if (n < HIDDEN_N + OUTPUT_N) {
      int o = n - HIDDEN_N;
      if (k < HIDDEN_N)                 v = Who_w[o * HIDDEN_N + k];      // h @ W_ho^T
      else if (k == 56)                 v = Who_b[o];
    }
    sW[idx] = f2bf(v);
  }

  // ---- Init augmented rows: h0 and the constant [1 | 0...] tail ----
  for (int idx = lane; idx < 16 * HIDDEN_N; idx += 32) {
    int r = idx / HIDDEN_N, c = idx % HIDDEN_N;
    sA[r * 64 + c] = f2bf(h0[(size_t)(row0 + r) * HIDDEN_N + c]);
  }
  if (lane < 16) {
    sA[lane * 64 + 56] = 0x3F80;                 // bf16(1.0) -> bias multiplier
    for (int c = 57; c < 64; ++c) sA[lane * 64 + c] = 0;
  }

  // ---- Load B fragments of Wcat^T once; reused all T steps ----
  v16bf bw[4][2];
  #pragma unroll
  for (int j = 0; j < 4; ++j)
    #pragma unroll
    for (int kb = 0; kb < 2; ++kb)
      bw[j][kb] = load_frag(sW, 16 * j + lm, kb, lh);

  // x prefetch: lane -> (row = lane/2, half = lane&1), 3 floats each
  const int xr = lane >> 1, xh = lane & 1;
  const float* xptr = x + ((size_t)(row0 + xr) * TT) * INPUT_N + xh * 3;
  float x0 = xptr[0], x1 = xptr[1], x2 = xptr[2];
  const v8f zacc = {};

  for (int t = 0; t < TT; ++t) {
    // stage x_t into A cols 50..55
    {
      int b = xr * 64 + HIDDEN_N + xh * 3;
      sA[b] = f2bf(x0); sA[b + 1] = f2bf(x1); sA[b + 2] = f2bf(x2);
    }
    v16bf a0 = load_frag(sA, lm, 0, lh);
    v16bf a1 = load_frag(sA, lm, 1, lh);

    // branch-free prefetch of x_{t+1} (clamped; overlaps the WMMAs)
    {
      int tn = (t + 1 < TT) ? (t + 1) : t;
      const float* p = xptr + (size_t)tn * INPUT_N;
      x0 = p[0]; x1 = p[1]; x2 = p[2];
    }

    // [16,64] x [64,64]: 4 N-tiles, K=64 chained (2 WMMA each)
    v8f c[4];
    #pragma unroll
    for (int j = 0; j < 4; ++j) {
      c[j] = __builtin_amdgcn_wmma_f32_16x16x32_bf16(false, a0, false, bw[j][0],
                                                     (short)0, zacc, false, false);
      c[j] = __builtin_amdgcn_wmma_f32_16x16x32_bf16(false, a1, false, bw[j][1],
                                                     (short)0, c[j], false, false);
    }

    // Consume oldest WMMA results first (tiles 0..2: tanh -> bf16 -> A cols) so
    // the scheduler can cover the WMMA->VALU hazard window with useful work.
    #pragma unroll
    for (int j = 0; j < 3; ++j) {
      int col = 16 * j + lm;
      #pragma unroll
      for (int v = 0; v < 8; ++v)
        sA[(lh * 8 + v) * 64 + col] = f2bf(fast_tanh(c[j][v]));   // col < 48 < HIDDEN
    }

    // cols 50,51 of tile 3 = y_{t-1} (logits, one step delayed). Stage + flush.
    if (t > 0) {
      if (lm == 2 || lm == 3) {
        int o = lm - 2, rb = lh * 8, slot = (t - 1) & 63;
        #pragma unroll
        for (int v = 0; v < 8; ++v)
          sY[((rb + v) * 64 + slot) * 2 + o] = c[3][v];
      }
      if ((t & 63) == 0) {                       // flush 64 staged steps, coalesced
        int tb = t - 64;
        #pragma unroll 4
        for (int r = 0; r < 16; ++r) {
          v4f val = *(const v4f*)&sY[r * 128 + lane * 4];
          *(v4f*)&logits[((size_t)(row0 + r) * TT + tb) * OUTPUT_N + lane * 4] = val;
        }
      }
    }

    // tile 3: only cols 48,49 belong to h
    {
      int col = 48 + lm;
      #pragma unroll
      for (int v = 0; v < 8; ++v) {
        if (col < HIDDEN_N)
          sA[(lh * 8 + v) * 64 + col] = f2bf(fast_tanh(c[3][v]));
      }
    }
  }

  // ---- Epilogue: y_{T-1} needs one more pass over tile 3 ----
  {
    v16bf a0 = load_frag(sA, lm, 0, lh);
    v16bf a1 = load_frag(sA, lm, 1, lh);
    v8f c3 = __builtin_amdgcn_wmma_f32_16x16x32_bf16(false, a0, false, bw[3][0],
                                                     (short)0, zacc, false, false);
    c3 = __builtin_amdgcn_wmma_f32_16x16x32_bf16(false, a1, false, bw[3][1],
                                                 (short)0, c3, false, false);
    if (lm == 2 || lm == 3) {
      int o = lm - 2, rb = lh * 8, slot = (TT - 1) & 63;
      #pragma unroll
      for (int v = 0; v < 8; ++v)
        sY[((rb + v) * 64 + slot) * 2 + o] = c3[v];
    }
    int tb = TT - 64;                            // final flush
    #pragma unroll 4
    for (int r = 0; r < 16; ++r) {
      v4f val = *(const v4f*)&sY[r * 128 + lane * 4];
      *(v4f*)&logits[((size_t)(row0 + r) * TT + tb) * OUTPUT_N + lane * 4] = val;
    }
    // h_final
    for (int idx = lane; idx < 16 * HIDDEN_N; idx += 32) {
      int r = idx / HIDDEN_N, cc = idx % HIDDEN_N;
      hfin[(size_t)(row0 + r) * HIDDEN_N + cc] = bf2f(sA[r * 64 + cc]);
    }
  }
}

extern "C" void kernel_launch(void* const* d_in, const int* in_sizes, int n_in,
                              void* d_out, int out_size, void* d_ws, size_t ws_size,
                              hipStream_t stream) {
  (void)in_sizes; (void)n_in; (void)out_size; (void)d_ws; (void)ws_size;
  const float* x     = (const float*)d_in[0];
  const float* h0    = (const float*)d_in[1];
  const float* Wih_w = (const float*)d_in[2];
  const float* Wih_b = (const float*)d_in[3];
  const float* Whh   = (const float*)d_in[4];
  const float* Who_w = (const float*)d_in[5];
  const float* Who_b = (const float*)d_in[6];
  float* out = (float*)d_out;

  dim3 grid(BB / 16);   // 256 single-wave workgroups, 16 batch rows each
  dim3 block(32);
  hipLaunchKernelGGL(elman_wmma_kernel, grid, block, 0, stream,
                     x, h0, Wih_w, Wih_b, Whh, Who_w, Who_b, out);
}
